// CosineAttentionBlock_29403346108653
// MI455X (gfx1250) — matrix operand order repaired and checked by reading, not compiled, not run on hardware
//
#include <hip/hip_runtime.h>
#include <hip/hip_bf16.h>
#include <math.h>
#include <stdint.h>

typedef __bf16 bf16_t;
typedef __attribute__((ext_vector_type(16))) __bf16 v16bf;
typedef __attribute__((ext_vector_type(8)))  __bf16 v8bf;
typedef __attribute__((ext_vector_type(8)))  float  v8f;
typedef unsigned int u32x4 __attribute__((ext_vector_type(4)));
typedef int          i32x8 __attribute__((ext_vector_type(8)));
typedef int          v4i   __attribute__((ext_vector_type(4)));

#define ROWS 8192   // B*N = 2*4096
#define DIM  512
#define MLP  2048

// ---- CDNA5 feature knobs -------------------------------------------------
#define USE_TDM 1   // tensor_load_to_lds for GEMM tile staging

#if defined(__HIP_DEVICE_COMPILE__) && __has_builtin(__builtin_amdgcn_global_load_async_to_lds_b128)
#define HAVE_ASYNC 1
#else
#define HAVE_ASYNC 0
#endif

#if HAVE_ASYNC
// builtin signature (from hipcc diagnostic): (v4i* global, v4i* lds, Imm, Imm)
#define GLB_AS1(p) ((__attribute__((address_space(1))) v4i*)(uintptr_t)(const void*)(p))
#define LDS_AS3(p) ((__attribute__((address_space(3))) v4i*)(unsigned)(uintptr_t)(const void*)(p))
#define ASYNC_LOAD16(gp, lp) \
    __builtin_amdgcn_global_load_async_to_lds_b128(GLB_AS1(gp), LDS_AS3(lp), 0, 0)
#endif

__device__ __forceinline__ float bf2f(bf16_t x) {
    unsigned short u = __builtin_bit_cast(unsigned short, x);
    unsigned int v = ((unsigned int)u) << 16;
    return __builtin_bit_cast(float, v);
}
__device__ __forceinline__ bf16_t f2bf(float f) {
    unsigned int u = __builtin_bit_cast(unsigned int, f);
    unsigned int r = u + 0x7FFFu + ((u >> 16) & 1u);   // round-to-nearest-even
    unsigned short h = (unsigned short)(r >> 16);
    return __builtin_bit_cast(bf16_t, h);
}

// ---------------------------------------------------------------------------
// elementwise fp32 -> bf16 cast
// ---------------------------------------------------------------------------
__global__ __launch_bounds__(256) void cast_f2b_k(const float* __restrict__ in,
                                                  bf16_t* __restrict__ out, int n) {
    int i = blockIdx.x * 256 + threadIdx.x;
    if (i < n) out[i] = f2bf(in[i]);
}

// ---------------------------------------------------------------------------
// transpose + cast: in fp32 [Kdim, Ndim] -> out bf16 [Ndim, Kdim]
// ---------------------------------------------------------------------------
__global__ __launch_bounds__(256) void transpose_cast_k(const float* __restrict__ in,
                                                        bf16_t* __restrict__ out,
                                                        int Kdim, int Ndim) {
    __shared__ float tile[32][33];
    const int t = threadIdx.x;
    const int tx = t & 31, ty = t >> 5;            // ty: 0..7
    const int k0 = blockIdx.y * 32, n0 = blockIdx.x * 32;
    #pragma unroll
    for (int i = 0; i < 32; i += 8)
        tile[ty + i][tx] = in[(size_t)(k0 + ty + i) * Ndim + n0 + tx];
    __syncthreads();
    #pragma unroll
    for (int i = 0; i < 32; i += 8)
        out[(size_t)(n0 + ty + i) * Kdim + k0 + tx] = f2bf(tile[tx][ty + i]);
}

// ---------------------------------------------------------------------------
// LayerNorm over 512 cols, one block (256 thr) per row, bf16 output
// ---------------------------------------------------------------------------
__global__ __launch_bounds__(256) void ln_bf16_k(const float* __restrict__ x,
                                                 const float* __restrict__ g,
                                                 const float* __restrict__ b,
                                                 bf16_t* __restrict__ out) {
    const int row = blockIdx.x;
    const int t = threadIdx.x;
    const float* xr = x + (size_t)row * DIM;
    float a0 = xr[t], a1 = xr[t + 256];
    float s = a0 + a1, ss = a0 * a0 + a1 * a1;
    #pragma unroll
    for (int o = 16; o > 0; o >>= 1) {
        s  += __shfl_xor(s, o);
        ss += __shfl_xor(ss, o);
    }
    __shared__ float sh[16];
    const int w = t >> 5;
    if ((t & 31) == 0) { sh[w] = s; sh[8 + w] = ss; }
    __syncthreads();
    float S = 0.f, SS = 0.f;
    #pragma unroll
    for (int i = 0; i < 8; i++) { S += sh[i]; SS += sh[8 + i]; }
    const float mu = S * (1.f / DIM);
    const float var = SS * (1.f / DIM) - mu * mu;
    const float rs = rsqrtf(var + 1e-5f);
    bf16_t* orow = out + (size_t)row * DIM;
    orow[t]       = f2bf((a0 - mu) * rs * g[t]       + b[t]);
    orow[t + 256] = f2bf((a1 - mu) * rs * g[t + 256] + b[t + 256]);
}

// ---------------------------------------------------------------------------
// Per-head L2 normalize: fp32 [ROWS,512] -> bf16 [ROWS,512]; one wave / 64-group
// ---------------------------------------------------------------------------
__global__ __launch_bounds__(256) void l2norm_k(const float* __restrict__ in,
                                                bf16_t* __restrict__ out) {
    const int t = threadIdx.x;
    const int lane = t & 31, w = t >> 5;
    const int g = blockIdx.x * 8 + w;
    const int row = g >> 3, h = g & 7;
    const size_t off = (size_t)row * DIM + h * 64;
    float a0 = in[off + lane], a1 = in[off + lane + 32];
    float ss = a0 * a0 + a1 * a1;
    #pragma unroll
    for (int o = 16; o > 0; o >>= 1) ss += __shfl_xor(ss, o);
    const float inv = 1.f / fmaxf(sqrtf(ss), 1e-12f);
    out[off + lane]      = f2bf(a0 * inv);
    out[off + lane + 32] = f2bf(a1 * inv);
}

// ---------------------------------------------------------------------------
// TDM: one tensor_load_to_lds moves a 128x32 bf16 tile (8KB), rows strided by
// row_stride elems. D# built per CDNA5 ISA ch.8 (2-D, data_size=2B, no pad).
// ---------------------------------------------------------------------------
#if USE_TDM
__device__ __forceinline__ void tdm_load_tile_128x32(unsigned lds_off,
                                                     const bf16_t* gsrc,
                                                     int row_stride) {
    const unsigned long long ga = (unsigned long long)(uintptr_t)gsrc;
    u32x4 g0;
    g0[0] = 1u;                                            // count=1, user desc
    g0[1] = lds_off;                                       // lds_addr (bytes)
    g0[2] = (unsigned)ga;                                  // global_addr[31:0]
    g0[3] = (unsigned)((ga >> 32) & 0x01FFFFFFull) | (2u << 30);  // addr hi | type=2
    i32x8 g1;
    g1[0] = 0x00010000;                // wg_mask=0, data_size=1 (2 bytes)
    g1[1] = (int)(32u << 16);          // tensor_dim0 = 32 (lo16 in [31:16])
    g1[2] = (int)(128u << 16);         // td0 hi=0 | tensor_dim1 = 128
    g1[3] = (int)(32u << 16);          // td1 hi=0 | tile_dim0 = 32
    g1[4] = 128;                       // tile_dim1 = 128, tile_dim2 = 0
    g1[5] = row_stride;                // tensor_dim0_stride[31:0] (elements)
    g1[6] = 0;
    g1[7] = 0;
    asm volatile("tensor_load_to_lds %0, %1" : : "s"(g0), "s"(g1) : "memory");
}
#endif

// ---------------------------------------------------------------------------
// WMMA inner compute: 8x v_wmma_f32_16x16x32_bf16 per wave per K-step.
// As: [128][32] row-major; Bs: [128 n][32 k] row-major (K-contiguous).
// ---------------------------------------------------------------------------
#define BMT 128
#define BNT 128
#define BKT 32

__device__ __forceinline__ void wmma_tile_compute(const bf16_t* Asb, const bf16_t* Bsb,
                                                  int wm, int wn, int l16, int lhi,
                                                  v8f (&acc)[2][4]) {
    v16bf af[2];
    #pragma unroll
    for (int mt = 0; mt < 2; mt++) {
        const bf16_t* ap = &Asb[(wm * 32 + mt * 16 + l16) * BKT + lhi * 8];
        v8bf lo = *(const v8bf*)ap;          // k = lhi*8 .. +7
        v8bf hi = *(const v8bf*)(ap + 16);   // k = 16+lhi*8 .. +7
        #pragma unroll
        for (int i = 0; i < 8; i++) { af[mt][i] = lo[i]; af[mt][8 + i] = hi[i]; }
    }
    #pragma unroll
    for (int nt = 0; nt < 4; nt++) {
        const bf16_t* bp = &Bsb[(wn * 64 + nt * 16 + l16) * BKT + lhi * 8];
        v8bf lo = *(const v8bf*)bp;          // k = lhi*8 .. +7
        v8bf hi = *(const v8bf*)(bp + 16);   // k = 16+lhi*8 .. +7
        v16bf bfm;
        #pragma unroll
        for (int i = 0; i < 8; i++) { bfm[i] = lo[i]; bfm[8 + i] = hi[i]; }
        #pragma unroll
        for (int mt = 0; mt < 2; mt++)
            acc[mt][nt] = __builtin_amdgcn_wmma_f32_16x16x32_bf16(
                false, af[mt], false, bfm, (short)0, acc[mt][nt], false, false);
    }
}

// ---------------------------------------------------------------------------
// Generic bf16 WMMA GEMM: C = A[MxK] * Bt^T, Bt is [N,K] (pre-transposed B).
// Block 256 thr = 8 waves, tile 128x128x32, double-buffered LDS.
// FLAGS: 1=+bias(col)  2=+resid(fp32)  4=GELU(exact)  8=store fp32  16=store bf16
// ---------------------------------------------------------------------------
template <int FLAGS>
__global__ __launch_bounds__(256) void gemm_bf16_k(
    const bf16_t* __restrict__ A, const bf16_t* __restrict__ Bt,
    const float* __restrict__ bias, const float* __restrict__ resid,
    float* __restrict__ outF, bf16_t* __restrict__ outB,
    int M, int N, int K) {
    (void)M;
    __shared__ __align__(16) bf16_t As[2][BMT * BKT];
    __shared__ __align__(16) bf16_t Bs[2][BNT * BKT];

    const int t = threadIdx.x;
    const int bm = blockIdx.y * BMT;
    const int bn = blockIdx.x * BNT;
    const int lane = t & 31, l16 = lane & 15, lhi = lane >> 4;
    const int w = t >> 5, wm = w & 3, wn = w >> 2;
    const int nsteps = K / BKT;

    v8f acc[2][4];
    #pragma unroll
    for (int mt = 0; mt < 2; mt++)
        #pragma unroll
        for (int nt = 0; nt < 4; nt++)
            #pragma unroll
            for (int r = 0; r < 8; r++) acc[mt][nt][r] = 0.f;

#if USE_TDM
    const unsigned ldsA0 = (unsigned)(uintptr_t)&As[0][0];
    const unsigned ldsA1 = (unsigned)(uintptr_t)&As[1][0];
    const unsigned ldsB0 = (unsigned)(uintptr_t)&Bs[0][0];
    const unsigned ldsB1 = (unsigned)(uintptr_t)&Bs[1][0];
    const bf16_t* Abase = A  + (size_t)bm * K;
    const bf16_t* Bbase = Bt + (size_t)bn * K;
    if (w == 0) {
        tdm_load_tile_128x32(ldsA0, Abase, K);
        tdm_load_tile_128x32(ldsB0, Bbase, K);
    }
    for (int i = 0; i < nsteps; ++i) {
        const int cur = i & 1;
        if (w == 0) {
            if (i + 1 < nsteps) {
                const int k1 = (i + 1) * BKT;
                tdm_load_tile_128x32(cur ? ldsA0 : ldsA1, Abase + k1, K);
                tdm_load_tile_128x32(cur ? ldsB0 : ldsB1, Bbase + k1, K);
                asm volatile("s_wait_tensorcnt 0x2" ::: "memory");
            } else {
                asm volatile("s_wait_tensorcnt 0x0" ::: "memory");
            }
        }
        __syncthreads();                       // tile i ready for all waves
        wmma_tile_compute(As[cur], Bs[cur], wm, wn, l16, lhi, acc);
        __syncthreads();                       // done before buffer reuse
    }
#else
    // sync staging fallback, double-buffered; both tiles are 128x32 contiguous
    const int arow = t >> 1;                   // 0..127
    const int acol = (t & 1) * 16;             // 0,16
    const bf16_t* Ag = A  + (size_t)(bm + arow) * K + acol;
    const bf16_t* Bg = Bt + (size_t)(bn + arow) * K + acol;
    {
        uint4 a0 = *(const uint4*)Ag,       a1 = *(const uint4*)(Ag + 8);
        uint4 b0 = *(const uint4*)Bg,       b1 = *(const uint4*)(Bg + 8);
        *(uint4*)&As[0][arow * BKT + acol]     = a0;
        *(uint4*)&As[0][arow * BKT + acol + 8] = a1;
        *(uint4*)&Bs[0][arow * BKT + acol]     = b0;
        *(uint4*)&Bs[0][arow * BKT + acol + 8] = b1;
    }
    for (int i = 0; i < nsteps; ++i) {
        const int cur = i & 1;
        uint4 a0, a1, b0, b1;
        if (i + 1 < nsteps) {
            const int k1 = (i + 1) * BKT;
            a0 = *(const uint4*)(Ag + k1);  a1 = *(const uint4*)(Ag + k1 + 8);
            b0 = *(const uint4*)(Bg + k1);  b1 = *(const uint4*)(Bg + k1 + 8);
        }
        __syncthreads();
        wmma_tile_compute(As[cur], Bs[cur], wm, wn, l16, lhi, acc);
        __syncthreads();
        if (i + 1 < nsteps) {
            *(uint4*)&As[1 - cur][arow * BKT + acol]     = a0;
            *(uint4*)&As[1 - cur][arow * BKT + acol + 8] = a1;
            *(uint4*)&Bs[1 - cur][arow * BKT + acol]     = b0;
            *(uint4*)&Bs[1 - cur][arow * BKT + acol + 8] = b1;
        }
    }
#endif

    // Epilogue. C layout: VGPR r -> row = base + r + lhi*8, col = base + l16
    #pragma unroll
    for (int mt = 0; mt < 2; mt++) {
        #pragma unroll
        for (int nt = 0; nt < 4; nt++) {
            const int col = bn + wn * 64 + nt * 16 + l16;
            const int rbase = bm + wm * 32 + mt * 16 + lhi * 8;
            float bv = 0.f;
            if constexpr (FLAGS & 1) bv = bias[col];
            #pragma unroll
            for (int r = 0; r < 8; r++) {
                const int row = rbase + r;
                float v = acc[mt][nt][r];
                if constexpr (FLAGS & 1) v += bv;
                if constexpr (FLAGS & 2) v += resid[(size_t)row * N + col];
                if constexpr (FLAGS & 4) v = 0.5f * v * (1.f + erff(v * 0.70710678118654752f));
                if constexpr (FLAGS & 8)  outF[(size_t)row * N + col] = v;
                if constexpr (FLAGS & 16) outB[(size_t)row * N + col] = f2bf(v);
            }
        }
    }
}

// ---------------------------------------------------------------------------
// S[b,h] = Kn^T * V  (64x64 per head), split-K over grid.y, fp32 atomics
// ---------------------------------------------------------------------------
__global__ __launch_bounds__(256) void kv_outer_k(const bf16_t* __restrict__ Kn,
                                                  const bf16_t* __restrict__ V,
                                                  float* __restrict__ S) {
    __shared__ __align__(16) bf16_t Ks[32 * 64];
    __shared__ __align__(16) bf16_t Vs[32 * 64];
    const int t = threadIdx.x;
    const int bh = blockIdx.x;
    const int b = bh >> 3, h = bh & 7;
    const int rbase = b * 4096 + blockIdx.y * 128;
    const int lrow = t >> 3;
    const int lcol = (t & 7) * 8;
    const int d = t >> 2;
    const int jb = (t & 3) * 16;
    float acc[16];
    #pragma unroll
    for (int i = 0; i < 16; i++) acc[i] = 0.f;
    for (int n0 = 0; n0 < 128; n0 += 32) {
        const size_t go = (size_t)(rbase + n0 + lrow) * DIM + h * 64 + lcol;
#if HAVE_ASYNC
        __syncthreads();
        ASYNC_LOAD16(Kn + go, &Ks[lrow * 64 + lcol]);
        ASYNC_LOAD16(V + go,  &Vs[lrow * 64 + lcol]);
        asm volatile("s_wait_asynccnt 0x0" ::: "memory");
        __syncthreads();
#else
        uint4 kv = *(const uint4*)(Kn + go);
        uint4 vv = *(const uint4*)(V + go);
        __syncthreads();
        *(uint4*)&Ks[lrow * 64 + lcol] = kv;
        *(uint4*)&Vs[lrow * 64 + lcol] = vv;
        __syncthreads();
#endif
        #pragma unroll 4
        for (int n = 0; n < 32; n++) {
            const float kd = bf2f(Ks[n * 64 + d]);
            #pragma unroll
            for (int jj = 0; jj < 16; jj++)
                acc[jj] += kd * bf2f(Vs[n * 64 + jb + jj]);
        }
    }
    float* Sp = S + (size_t)bh * 4096 + d * 64 + jb;
    #pragma unroll
    for (int jj = 0; jj < 16; jj++) atomicAdd(Sp + jj, acc[jj]);
}

// ---------------------------------------------------------------------------
// out_inner[row, h*64+j] = sum_d Qn[row, h*64+d] * S[b,h,d,j]
// ---------------------------------------------------------------------------
__global__ __launch_bounds__(256) void qs_k(const bf16_t* __restrict__ Qn,
                                            const bf16_t* __restrict__ S,
                                            bf16_t* __restrict__ out) {
    __shared__ __align__(16) bf16_t Ss[64 * 64];
    __shared__ __align__(16) bf16_t Qs[64 * 64];
    const int t = threadIdx.x;
    const int rowbase = blockIdx.x * 64;
    const int h = blockIdx.y;
    const int b = rowbase >> 12;
    const int bh = b * 8 + h;
    const int rl = t >> 2;
    const int d0 = (t & 3) * 16;
    const bf16_t* sp = S + (size_t)bh * 4096 + t * 16;
    const bf16_t* qp = Qn + (size_t)(rowbase + rl) * DIM + h * 64 + d0;
#if HAVE_ASYNC
    ASYNC_LOAD16(sp,     &Ss[t * 16]);
    ASYNC_LOAD16(sp + 8, &Ss[t * 16 + 8]);
    ASYNC_LOAD16(qp,     &Qs[rl * 64 + d0]);
    ASYNC_LOAD16(qp + 8, &Qs[rl * 64 + d0 + 8]);
    asm volatile("s_wait_asynccnt 0x0" ::: "memory");
    __syncthreads();
#else
    *(uint4*)&Ss[t * 16]         = *(const uint4*)sp;
    *(uint4*)&Ss[t * 16 + 8]     = *(const uint4*)(sp + 8);
    *(uint4*)&Qs[rl * 64 + d0]     = *(const uint4*)qp;
    *(uint4*)&Qs[rl * 64 + d0 + 8] = *(const uint4*)(qp + 8);
    __syncthreads();
#endif
    const int jb = (t & 3) * 16;
    float acc[16];
    #pragma unroll
    for (int i = 0; i < 16; i++) acc[i] = 0.f;
    #pragma unroll 4
    for (int d = 0; d < 64; d++) {
        const float qd = bf2f(Qs[rl * 64 + d]);
        #pragma unroll
        for (int jj = 0; jj < 16; jj++)
            acc[jj] += qd * bf2f(Ss[d * 64 + jb + jj]);
    }
    bf16_t* op = out + (size_t)(rowbase + rl) * DIM + h * 64 + jb;
    #pragma unroll
    for (int jj = 0; jj < 16; jj++) op[jj] = f2bf(acc[jj]);
}

// ---------------------------------------------------------------------------
// host
// ---------------------------------------------------------------------------
static inline char* bump(char*& p, size_t bytes) {
    char* r = p;
    p += (bytes + 255) & ~(size_t)255;
    return r;
}

extern "C" void kernel_launch(void* const* d_in, const int* in_sizes, int n_in,
                              void* d_out, int out_size, void* d_ws, size_t ws_size,
                              hipStream_t stream) {
    (void)in_sizes; (void)n_in; (void)out_size; (void)ws_size;
    const float* q   = (const float*)d_in[0];
    const float* k   = (const float*)d_in[1];
    const float* v   = (const float*)d_in[2];
    const float* Wq  = (const float*)d_in[3];
    const float* Wk  = (const float*)d_in[4];
    const float* Wv  = (const float*)d_in[5];
    const float* Wo  = (const float*)d_in[6];
    const float* bo  = (const float*)d_in[7];
    const float* g1  = (const float*)d_in[8];
    const float* b1  = (const float*)d_in[9];
    const float* g2  = (const float*)d_in[10];
    const float* b2  = (const float*)d_in[11];
    const float* W1  = (const float*)d_in[12];
    const float* bf1 = (const float*)d_in[13];
    const float* W2  = (const float*)d_in[14];
    const float* bf2 = (const float*)d_in[15];
    float* out = (float*)d_out;

    char* p = (char*)d_ws;
    bf16_t* lnq_b = (bf16_t*)bump(p, (size_t)ROWS * DIM * 2);
    bf16_t* kb    = (bf16_t*)bump(p, (size_t)ROWS * DIM * 2);
    bf16_t* vbin  = (bf16_t*)bump(p, (size_t)ROWS * DIM * 2);
    bf16_t* Wqt   = (bf16_t*)bump(p, (size_t)DIM * DIM * 2);   // [N,K] transposed
    bf16_t* Wkt   = (bf16_t*)bump(p, (size_t)DIM * DIM * 2);
    bf16_t* Wvt   = (bf16_t*)bump(p, (size_t)DIM * DIM * 2);
    bf16_t* Wot   = (bf16_t*)bump(p, (size_t)DIM * DIM * 2);
    bf16_t* W1t   = (bf16_t*)bump(p, (size_t)DIM * MLP * 2);   // [2048,512]
    bf16_t* W2t   = (bf16_t*)bump(p, (size_t)MLP * DIM * 2);   // [512,2048]
    float*  Qf    = (float*)bump(p, (size_t)ROWS * DIM * 4);
    float*  Kf    = (float*)bump(p, (size_t)ROWS * DIM * 4);
    bf16_t* Vb    = (bf16_t*)bump(p, (size_t)ROWS * DIM * 2);
    bf16_t* Qn    = (bf16_t*)bump(p, (size_t)ROWS * DIM * 2);
    bf16_t* Kn    = (bf16_t*)bump(p, (size_t)ROWS * DIM * 2);
    float*  Sf    = (float*)bump(p, (size_t)16 * 64 * 64 * 4);
    bf16_t* Sb    = (bf16_t*)bump(p, (size_t)16 * 64 * 64 * 2);
    bf16_t* attnb = (bf16_t*)bump(p, (size_t)ROWS * DIM * 2);
    float*  x     = (float*)bump(p, (size_t)ROWS * DIM * 4);
    bf16_t* lnx_b = (bf16_t*)bump(p, (size_t)ROWS * DIM * 2);
    bf16_t* hb    = (bf16_t*)bump(p, (size_t)ROWS * MLP * 2);

    const int nRD = ROWS * DIM;
    // input casts
    cast_f2b_k<<<(nRD + 255) / 256, 256, 0, stream>>>(k, kb, nRD);
    cast_f2b_k<<<(nRD + 255) / 256, 256, 0, stream>>>(v, vbin, nRD);
    // weight transpose+cast: W[K,N] -> Wt[N,K]
    transpose_cast_k<<<dim3(DIM / 32, DIM / 32), 256, 0, stream>>>(Wq, Wqt, DIM, DIM);
    transpose_cast_k<<<dim3(DIM / 32, DIM / 32), 256, 0, stream>>>(Wk, Wkt, DIM, DIM);
    transpose_cast_k<<<dim3(DIM / 32, DIM / 32), 256, 0, stream>>>(Wv, Wvt, DIM, DIM);
    transpose_cast_k<<<dim3(DIM / 32, DIM / 32), 256, 0, stream>>>(Wo, Wot, DIM, DIM);
    transpose_cast_k<<<dim3(MLP / 32, DIM / 32), 256, 0, stream>>>(W1, W1t, DIM, MLP);
    transpose_cast_k<<<dim3(DIM / 32, MLP / 32), 256, 0, stream>>>(W2, W2t, MLP, DIM);
    // LN(q)
    ln_bf16_k<<<ROWS, 256, 0, stream>>>(q, g1, b1, lnq_b);

    // projections
    dim3 g512(DIM / BNT, ROWS / BMT);
    gemm_bf16_k<8><<<g512, 256, 0, stream>>>(lnq_b, Wqt, nullptr, nullptr, Qf, nullptr, ROWS, DIM, DIM);
    gemm_bf16_k<8><<<g512, 256, 0, stream>>>(kb,    Wkt, nullptr, nullptr, Kf, nullptr, ROWS, DIM, DIM);
    gemm_bf16_k<16><<<g512, 256, 0, stream>>>(vbin, Wvt, nullptr, nullptr, nullptr, Vb, ROWS, DIM, DIM);

    // cosine normalize
    l2norm_k<<<ROWS, 256, 0, stream>>>(Qf, Qn);
    l2norm_k<<<ROWS, 256, 0, stream>>>(Kf, Kn);

    // linear-attention reassociation: S = Kn^T V, then out_inner = Qn S
    (void)hipMemsetAsync(Sf, 0, (size_t)16 * 64 * 64 * 4, stream);
    dim3 gkv(16, 32);
    kv_outer_k<<<gkv, 256, 0, stream>>>(Kn, Vb, Sf);
    cast_f2b_k<<<(16 * 64 * 64 + 255) / 256, 256, 0, stream>>>(Sf, Sb, 16 * 64 * 64);
    dim3 gqs(ROWS / 64, 8);
    qs_k<<<gqs, 256, 0, stream>>>(Qn, Sb, attnb);

    // x = attn_inner @ Wo + bo + q
    gemm_bf16_k<1 | 2 | 8><<<g512, 256, 0, stream>>>(attnb, Wot, bo, q, x, nullptr, ROWS, DIM, DIM);
    // LN(x)
    ln_bf16_k<<<ROWS, 256, 0, stream>>>(x, g2, b2, lnx_b);
    // h = gelu(lnx @ W1 + bf1)
    dim3 gmlp(MLP / BNT, ROWS / BMT);
    gemm_bf16_k<1 | 4 | 16><<<gmlp, 256, 0, stream>>>(lnx_b, W1t, bf1, nullptr, nullptr, hb, ROWS, MLP, DIM);
    // out = h @ W2 + bf2 + x
    gemm_bf16_k<1 | 2 | 8><<<g512, 256, 0, stream>>>(hb, W2t, bf2, x, out, nullptr, ROWS, DIM, MLP);
}
